// LSTMModelCustom_22196390985729
// MI455X (gfx1250) — matrix operand order repaired and checked by reading, not compiled, not run on hardware
//
#include <hip/hip_runtime.h>
#include <math.h>

typedef __bf16 v16bf  __attribute__((ext_vector_type(16)));
typedef __bf16 bf16x8 __attribute__((ext_vector_type(8)));
typedef float  v8f    __attribute__((ext_vector_type(8)));

#define INPUT_DIM 128
#define HIDDEN    512
#define NGATES    2048   // 4*HIDDEN
#define KDIM      640    // INPUT_DIM + HIDDEN
#define BATCH     64
#define SEQ       1024

// ---------------- prep kernels ----------------

__global__ void k_convert_x(const float* __restrict__ x, __bf16* __restrict__ xb, int n) {
    int i = blockIdx.x * blockDim.x + threadIdx.x;
    if (i < n) xb[i] = (__bf16)x[i];
}

// Build combined transposed weights: Wc[n][k] (2048 x 640), k<128 from W, else from U.
// Transposed so B-fragments (fixed column N, varying K) are contiguous loads.
__global__ void k_build_w(const float* __restrict__ W, const float* __restrict__ U,
                          __bf16* __restrict__ Wc) {
    int i = blockIdx.x * blockDim.x + threadIdx.x;
    if (i >= NGATES * KDIM) return;
    int n = i / KDIM, k = i % KDIM;
    float v = (k < INPUT_DIM) ? W[(size_t)k * NGATES + n]
                              : U[(size_t)(k - INPUT_DIM) * NGATES + n];
    Wc[i] = (__bf16)v;
}

__global__ void k_zero(__bf16* __restrict__ hb, float* __restrict__ c) {
    int i = blockIdx.x * blockDim.x + threadIdx.x;
    if (i < 2 * BATCH * HIDDEN) hb[i] = (__bf16)0.0f;
    if (i < BATCH * HIDDEN)     c[i]  = 0.0f;
}

// ---------------- one LSTM timestep ----------------
// grid = 32 WGs (one per 16 hidden cols), block = 512 (16 waves).
// wave w: row-tile mt = w&3, gate = w>>2. GEMM: gates(64x2048) = [x_t;h] @ Wc^T + bias
// (K = 640 = 20 bf16 WMMA steps), then fused cell update on this WG's column slice.

__launch_bounds__(512)
__global__ void k_lstm_step(const __bf16* __restrict__ xb,     // (64,1024,128) bf16
                            const __bf16* __restrict__ Wc,     // (2048,640) bf16
                            const float*  __restrict__ bias,   // (2048,) f32
                            const __bf16* __restrict__ h_in,   // (64,512) bf16
                            __bf16*       __restrict__ h_out,  // (64,512) bf16
                            float*        __restrict__ c_st,   // (64,512) f32
                            float*        __restrict__ h_f32,  // (64,512) f32
                            int s) {
    __shared__ float lds[4][64][17];   // [gate][row][col], padded

    const int tid    = threadIdx.x;
    const int wave   = tid >> 5;
    const int lane   = tid & 31;
    const int half   = lane >> 4;
    const int idx16  = lane & 15;
    const int mt     = wave & 3;          // row tile (batch)
    const int gate   = wave >> 2;         // 0..3
    const int hc_base = blockIdx.x << 4;  // hidden column slice base
    const int n_base  = gate * HIDDEN + hc_base;

    const int arow = (mt << 4) + idx16;   // A fragment: this lane's batch row
    const int ncol = n_base + idx16;      // B fragment: this lane's output column

    const float bv = bias[n_base + idx16];
    v8f acc = { bv, bv, bv, bv, bv, bv, bv, bv };

    const __bf16* xrow = xb + ((size_t)arow * SEQ + (size_t)s) * INPUT_DIM;
    const __bf16* hrow = h_in + (size_t)arow * HIDDEN;
    const __bf16* brow = Wc + (size_t)ncol * KDIM;

#pragma unroll
    for (int ks = 0; ks < KDIM / 32; ++ks) {
        const int kk = ks * 32;
        // K 0..127 comes from x_t, 128..639 from h_{t-1}; both K-contiguous rows.
        const __bf16* abase = (kk < INPUT_DIM) ? (xrow + kk) : (hrow + (kk - INPUT_DIM));
        bf16x8 alo = *(const bf16x8*)(abase + 8 * half);        // K = kk + 8*half + 0..7
        bf16x8 ahi = *(const bf16x8*)(abase + 16 + 8 * half);   // K = kk + 16 + 8*half + 0..7
        bf16x8 blo = *(const bf16x8*)(brow + kk + 8 * half);   // K = kk + 8*half + 0..7
        bf16x8 bhi = *(const bf16x8*)(brow + kk + 16 + 8 * half);
        v16bf afrag, bfrag;
#pragma unroll
        for (int j = 0; j < 8; ++j) {
            afrag[j] = alo[j]; afrag[j + 8] = ahi[j];
            bfrag[j] = blo[j]; bfrag[j + 8] = bhi[j];
        }
        acc = __builtin_amdgcn_wmma_f32_16x16x32_bf16(
                  false, afrag, false, bfrag, (short)0, acc, false, false);
    }

    // C layout: VGPR i -> row = i + 8*half, col = idx16
#pragma unroll
    for (int i = 0; i < 8; ++i)
        lds[gate][(mt << 4) + (half << 3) + i][idx16] = acc[i];

    __syncthreads();

    // Fused cell update on this WG's 64x16 slice (all 4 gates local in LDS).
    for (int e = tid; e < BATCH * 16; e += 512) {
        int r = e >> 4, cc = e & 15;
        float gi = lds[0][r][cc];
        float gf = lds[1][r][cc];
        float gg = lds[2][r][cc];
        float go = lds[3][r][cc];
        float it = 1.0f / (1.0f + expf(-gi));
        float ft = 1.0f / (1.0f + expf(-gf));
        float gt = tanhf(gg);
        float ot = 1.0f / (1.0f + expf(-go));
        int idx = r * HIDDEN + hc_base + cc;
        float cn = ft * c_st[idx] + it * gt;
        c_st[idx]  = cn;
        float h    = ot * tanhf(cn);
        h_f32[idx] = h;
        h_out[idx] = (__bf16)h;
    }
}

// ---------------- final FC on last hidden state ----------------
__global__ void k_fc(const float* __restrict__ h, const float* __restrict__ fw,
                     const float* __restrict__ fb, float* __restrict__ out) {
    int e = blockIdx.x * blockDim.x + threadIdx.x;
    if (e >= BATCH * 7) return;
    int r = e / 7, j = e % 7;
    float acc = fb[j];
    for (int k = 0; k < HIDDEN; ++k)
        acc += h[r * HIDDEN + k] * fw[k * 7 + j];
    out[e] = acc;
}

// ---------------- host entry ----------------
extern "C" void kernel_launch(void* const* d_in, const int* in_sizes, int n_in,
                              void* d_out, int out_size, void* d_ws, size_t ws_size,
                              hipStream_t stream) {
    (void)in_sizes; (void)n_in; (void)out_size; (void)ws_size;
    const float* x    = (const float*)d_in[0];
    const float* W    = (const float*)d_in[1];
    const float* U    = (const float*)d_in[2];
    const float* bias = (const float*)d_in[3];
    const float* fc_w = (const float*)d_in[4];
    const float* fc_b = (const float*)d_in[5];
    float* out = (float*)d_out;

    // Workspace layout (~19 MB total)
    char* ws = (char*)d_ws;
    __bf16* xb = (__bf16*)(ws);                    // 64*1024*128 bf16 = 16,777,216 B
    __bf16* Wc = (__bf16*)(ws + 16777216);         //  2048*640 bf16   =  2,621,440 B
    __bf16* hb = (__bf16*)(ws + 19398656);         //  2 * 64*512 bf16 =    131,072 B
    float*  c  = (float* )(ws + 19529728);         //  64*512 f32      =    131,072 B
    float*  hf = (float* )(ws + 19660800);         //  64*512 f32      =    131,072 B

    const int NX = BATCH * SEQ * INPUT_DIM;        // 8,388,608
    k_convert_x<<<(NX + 255) / 256, 256, 0, stream>>>(x, xb, NX);
    k_build_w<<<(NGATES * KDIM + 255) / 256, 256, 0, stream>>>(W, U, Wc);
    k_zero<<<(2 * BATCH * HIDDEN + 255) / 256, 256, 0, stream>>>(hb, c);

    for (int s = 0; s < SEQ; ++s) {
        const __bf16* hin  = hb + (size_t)(s & 1) * (BATCH * HIDDEN);
        __bf16*       hout = hb + (size_t)((s + 1) & 1) * (BATCH * HIDDEN);
        k_lstm_step<<<HIDDEN / 16, 512, 0, stream>>>(xb, Wc, bias, hin, hout, c, hf, s);
    }

    k_fc<<<(BATCH * 7 + 255) / 256, 256, 0, stream>>>(hf, fc_w, fc_b, out);
}
